// MultiRankTreeHead_22058952032970
// MI455X (gfx1250) — matrix operand order repaired and checked by reading, not compiled, not run on hardware
//
#include <hip/hip_runtime.h>
#include <hip/hip_bf16.h>
#include <vector>
#include <deque>
#include <array>

// ---------------------------------------------------------------------------
// Types for WMMA fragments (CDNA5 / gfx1250, wave32)
// ---------------------------------------------------------------------------
typedef __attribute__((ext_vector_type(16))) __bf16         v16bf;
typedef __attribute__((ext_vector_type(8)))  float          v8f;
typedef __attribute__((ext_vector_type(8)))  unsigned short us8;
typedef __attribute__((ext_vector_type(4)))  float          f4v;
typedef __attribute__((ext_vector_type(4)))  unsigned short us4;

union Frag {
    us8   u8[2];
    v16bf bf;
};

__device__ __forceinline__ unsigned short f2bf_rne(float f) {
    unsigned u = __float_as_uint(f);
    u += 0x7fffu + ((u >> 16) & 1u);
    return (unsigned short)(u >> 16);
}
__device__ __forceinline__ float bf2f(unsigned short h) {
    return __uint_as_float(((unsigned)h) << 16);
}
__device__ __forceinline__ float sigm(float x) {
    return 1.0f / (1.0f + expf(-x));
}
__device__ __forceinline__ float softplusf(float x) {
    return (x > 15.0f) ? x : log1pf(expf(x));
}

static constexpr int Bb   = 64;    // batch
static constexpr int Mh   = 2048;  // hidden
static constexpr int G4   = 8192;  // 4*M gates
static constexpr int NC   = 100;
static constexpr int NINT = NC - 1;   // 99 internal nodes
static constexpr int NNOD = 2 * NC - 1;

// ---------------------------------------------------------------------------
// K1: global average pool (B,C,7,7) -> bf16 (B,C)   (single-touch: NT loads)
// ---------------------------------------------------------------------------
__global__ void pool_kernel(const float* __restrict__ x,
                            unsigned short* __restrict__ pooled, int total) {
    int idx = blockIdx.x * blockDim.x + threadIdx.x;
    if (idx >= total) return;
    const float* p = x + (size_t)idx * 49;
    float s = 0.f;
#pragma unroll
    for (int i = 0; i < 49; ++i) s += __builtin_nontemporal_load(p + i);
    pooled[idx] = f2bf_rne(s * (1.0f / 49.0f));
}

// ---------------------------------------------------------------------------
// K2: fp32 -> bf16 weight convert (n multiple of 4; NT source loads)
// ---------------------------------------------------------------------------
__global__ void f2bf_kernel(const float* __restrict__ in,
                            unsigned short* __restrict__ out, int n4) {
    int i = blockIdx.x * blockDim.x + threadIdx.x;
    if (i >= n4) return;
    f4v v = __builtin_nontemporal_load(reinterpret_cast<const f4v*>(in) + i);
    us4 r;
    r.x = f2bf_rne(v.x); r.y = f2bf_rne(v.y);
    r.z = f2bf_rne(v.z); r.w = f2bf_rne(v.w);
    reinterpret_cast<us4*>(out)[i] = r;
}

// ---------------------------------------------------------------------------
// K3: C[64,N] = A[64,K](bf16) @ W[N,K](bf16)^T + bias1 (+bias2)
//     each wave: 64 x 16 tile (4 row-tiles stacked), block = 8 waves = 128 cols
//     ping-pong double buffer (no register copies, partial loadcnt waits)
// ---------------------------------------------------------------------------
__global__ void gemm_wmma_bias(const unsigned short* __restrict__ A,
                               const unsigned short* __restrict__ W,
                               const float* __restrict__ bias1,
                               const float* __restrict__ bias2,
                               float* __restrict__ outF,
                               unsigned short* __restrict__ outBF,
                               int N, int K) {
    const int wid  = threadIdx.x >> 5;
    const int lane = threadIdx.x & 31;
    const int nn   = lane & 15;   // B-col / A-row within tile
    const int kg   = lane >> 4;   // lane half
    const int col0 = blockIdx.x * 128 + wid * 16;

    v8f acc[4] = {v8f{}, v8f{}, v8f{}, v8f{}};
    const unsigned short* wrow = W + (size_t)(col0 + nn) * K;

    Frag a0[4], b0, a1[4], b1;
    auto loadAB = [&](int kt, Frag (&a)[4], Frag& b) {
        b.u8[0] = __builtin_nontemporal_load(
            reinterpret_cast<const us8*>(wrow + kt + kg * 16));
        b.u8[1] = __builtin_nontemporal_load(
            reinterpret_cast<const us8*>(wrow + kt + kg * 16 + 8));
#pragma unroll
        for (int rt = 0; rt < 4; ++rt) {
            const unsigned short* arow = A + (size_t)(rt * 16 + nn) * K + kt;
            a[rt].u8[0] = *reinterpret_cast<const us8*>(arow + kg * 16);
            a[rt].u8[1] = *reinterpret_cast<const us8*>(arow + kg * 16 + 8);
        }
    };
    auto mma = [&](Frag (&a)[4], Frag& b) {
#pragma unroll
        for (int rt = 0; rt < 4; ++rt)
            acc[rt] = __builtin_amdgcn_wmma_f32_16x16x32_bf16(
                false, a[rt].bf, false, b.bf, (short)0, acc[rt], false, false);
    };

    loadAB(0, a0, b0);
    int kt = 0;
    for (; kt < K - 64; kt += 64) {          // K multiple of 64
        loadAB(kt + 32, a1, b1);
        mma(a0, b0);
        loadAB(kt + 64, a0, b0);
        mma(a1, b1);
    }
    loadAB(kt + 32, a1, b1);
    mma(a0, b0);
    mma(a1, b1);

    const int col = col0 + nn;
    float bv = bias1[col] + (bias2 ? bias2[col] : 0.0f);
#pragma unroll
    for (int rt = 0; rt < 4; ++rt)
#pragma unroll
        for (int v = 0; v < 8; ++v) {
            int row = rt * 16 + v + 8 * kg;
            float val = acc[rt][v] + bv;
            if (outF)  outF[(size_t)row * N + col] = val;
            if (outBF) outBF[(size_t)row * N + col] = f2bf_rne(val);
        }
}

// ---------------------------------------------------------------------------
// K4: fused per-level kernel:
//   gates[64,8192] = xg + h_parent @ W_hh^T     (skipped for root: h_parent=0)
//   LSTM cell -> c[node] (f32), h[node] (bf16)
// Each wave: 64 batch rows x 16 hidden cols x 4 gates = 16 accumulators.
// h_parent chunks (64x32 bf16, 4 KB) staged in LDS (80 B padded row stride);
// W_hh B-tiles stream NT from L2, ping-pong double-buffered in registers.
// grid = (M/128, nodes_in_level), block = 256.
// ---------------------------------------------------------------------------
struct LevelDesc {
    int node[40];
    int parent[40];
    int count;
};

static constexpr int SROW = 40;  // padded LDS row stride in halfwords

__global__ void lstm_level_kernel(const unsigned short* __restrict__ Whh,  // [8192,2048] bf16
                                  const float* __restrict__ xg,            // [64,8192]
                                  unsigned short* __restrict__ h_all,      // [99,64,2048] bf16
                                  float* __restrict__ c_all,               // [99,64,2048]
                                  LevelDesc ld, int isRoot) {
    __shared__ unsigned short sA[64 * SROW];   // 5120 B

    const int wid  = threadIdx.x >> 5;
    const int lane = threadIdx.x & 31;
    const int nn   = lane & 15;
    const int kg   = lane >> 4;
    const int node   = ld.node[blockIdx.y];
    const int parent = ld.parent[blockIdx.y];
    const int m0 = blockIdx.x * 128 + wid * 16;

    v8f acc[4][4];
#pragma unroll
    for (int rt = 0; rt < 4; ++rt)
#pragma unroll
        for (int gi = 0; gi < 4; ++gi) acc[rt][gi] = v8f{};

    if (!isRoot) {
        const unsigned short* hp = h_all + (size_t)parent * Bb * Mh;
        const int srow = threadIdx.x >> 2;       // 0..63  (batch row)
        const int sq   = (threadIdx.x & 3) * 8;  // halfword chunk within 32

        // prime: stage chunk kt=0
        *reinterpret_cast<us8*>(sA + srow * SROW + sq) =
            *reinterpret_cast<const us8*>(hp + (size_t)srow * Mh + sq);
        __syncthreads();

        Frag bcur[4], bnext[4];
        auto loadB = [&](int kt, Frag (&bf)[4]) {
#pragma unroll
            for (int gi = 0; gi < 4; ++gi) {
                const unsigned short* wr =
                    Whh + (size_t)(gi * Mh + m0 + nn) * Mh + kt + kg * 16;
                bf[gi].u8[0] = __builtin_nontemporal_load(
                    reinterpret_cast<const us8*>(wr));
                bf[gi].u8[1] = __builtin_nontemporal_load(
                    reinterpret_cast<const us8*>(wr + 8));
            }
        };
        loadB(0, bcur);

        auto step = [&](int it, Frag (&bc)[4], Frag (&bn)[4]) {
            const int kt = it * 32;
            const bool more = it < (Mh / 32 - 1);

            // A fragments from LDS (shared across the 8 waves)
            Frag a[4];
#pragma unroll
            for (int rt = 0; rt < 4; ++rt) {
                const unsigned short* lr = sA + (rt * 16 + nn) * SROW;
                a[rt].u8[0] = *reinterpret_cast<const us8*>(lr + kg * 16);
                a[rt].u8[1] = *reinterpret_cast<const us8*>(lr + kg * 16 + 8);
            }

            // issue next-iteration global loads early (overlap with WMMAs)
            us8 stageNext;
            if (more) {
                stageNext = *reinterpret_cast<const us8*>(
                    hp + (size_t)srow * Mh + kt + 32 + sq);
                loadB(kt + 32, bn);
            }

#pragma unroll
            for (int gi = 0; gi < 4; ++gi)
#pragma unroll
                for (int rt = 0; rt < 4; ++rt)
                    acc[rt][gi] = __builtin_amdgcn_wmma_f32_16x16x32_bf16(
                        false, a[rt].bf, false, bc[gi].bf, (short)0,
                        acc[rt][gi], false, false);

            if (more) {
                __syncthreads();   // everyone done reading sA chunk kt
                *reinterpret_cast<us8*>(sA + srow * SROW + sq) = stageNext;
                __syncthreads();   // chunk kt+32 visible
            }
        };

        for (int it = 0; it < Mh / 32; it += 2) {   // 64 steps, even
            step(it,     bcur, bnext);
            step(it + 1, bnext, bcur);
        }
    }

    const int mcol = m0 + nn;
    const size_t outBase = (size_t)node * Bb * Mh;
    const size_t parBase = (size_t)parent * Bb * Mh;
#pragma unroll
    for (int rt = 0; rt < 4; ++rt)
#pragma unroll
        for (int v = 0; v < 8; ++v) {
            int b = rt * 16 + v + 8 * kg;
            const float* xgr = xg + (size_t)b * G4 + mcol;
            float ig = acc[rt][0][v] + xgr[0];
            float fg = acc[rt][1][v] + xgr[Mh];
            float gg = acc[rt][2][v] + xgr[2 * Mh];
            float og = acc[rt][3][v] + xgr[3 * Mh];
            float cp = isRoot ? 0.0f : c_all[parBase + (size_t)b * Mh + mcol];
            float cn = sigm(fg) * cp + sigm(ig) * tanhf(gg);
            float hn = sigm(og) * tanhf(cn);
            c_all[outBase + (size_t)b * Mh + mcol] = cn;
            h_all[outBase + (size_t)b * Mh + mcol] = f2bf_rne(hn);
        }
}

// ---------------------------------------------------------------------------
// K5: o[n,b] = dot(h[n,b,:], W_score[n,:]) + b_score[n]   (one wave per pair)
// ---------------------------------------------------------------------------
__global__ void score_kernel(const unsigned short* __restrict__ h_all,
                             const float* __restrict__ Wsc,
                             const float* __restrict__ bsc,
                             float* __restrict__ o) {
    const int wid  = threadIdx.x >> 5;
    const int lane = threadIdx.x & 31;
    const int pair = blockIdx.x * 8 + wid;
    if (pair >= NINT * Bb) return;
    const int n = pair >> 6;
    const int b = pair & 63;
    const unsigned short* hr = h_all + ((size_t)n * Bb + b) * Mh;
    const float* wr = Wsc + (size_t)n * Mh;
    float s = 0.f;
    for (int m = lane; m < Mh; m += 32) s += bf2f(hr[m]) * wr[m];
#pragma unroll
    for (int off = 16; off > 0; off >>= 1) s += __shfl_down(s, off, 32);
    if (lane == 0) o[n * Bb + b] = s + bsc[n];
}

// ---------------------------------------------------------------------------
// K6: leaf logits: walk root path, sum softplus(+/- o)
// ---------------------------------------------------------------------------
struct TreeDesc {
    int fa[NNOD];
    int isLeft[NNOD];
};

__global__ void logit_kernel(const float* __restrict__ o,
                             float* __restrict__ out, TreeDesc td) {
    int idx = blockIdx.x * blockDim.x + threadIdx.x;
    if (idx >= Bb * NC) return;
    int b = idx / NC, cls = idx % NC;
    int node = cls + NINT;
    float s = 1.0f;
    while (node != 0) {
        int p = td.fa[node];
        float ov = o[p * Bb + b];
        s += softplusf(td.isLeft[node] ? ov : -ov);
        node = p;
    }
    out[(size_t)b * NC + cls] = s;
}

// ---------------------------------------------------------------------------
// Host: faithful tree build (matches reference build_tree) + launches
// ---------------------------------------------------------------------------
static void build_tree_host(int* fa, int* isLeft) {
    std::vector<std::array<int, 2>> interval(NNOD, {0, NC - 1});
    for (int i = 0; i < NNOD; ++i) { fa[i] = 0; isLeft[i] = 0; }
    int idx_node = 1;
    std::deque<int> q;
    q.push_back(0);
    while (!q.empty()) {
        int x = q.front(); q.pop_front();
        int l = interval[x][0], r = interval[x][1];
        if (r - l == 0) continue;
        int il, ir;
        if (r - l == 1)      { il = l + NC - 1; ir = r + NC - 1; }
        else if (r - l == 2) { il = idx_node;   ir = r + NC - 1; idx_node += 1; }
        else                 { il = idx_node;   ir = idx_node + 1; idx_node += 2; }
        q.push_back(il); q.push_back(ir);
        fa[il] = x; fa[ir] = x;
        isLeft[il] = 1; isLeft[ir] = 0;
        interval[il] = {l, (l + r) / 2};
        interval[ir] = {(l + r) / 2 + 1, r};
    }
}

extern "C" void kernel_launch(void* const* d_in, const int* in_sizes, int n_in,
                              void* d_out, int out_size, void* d_ws, size_t ws_size,
                              hipStream_t stream) {
    const float* x       = (const float*)d_in[0];
    const float* W_fc    = (const float*)d_in[1];
    const float* b_fc    = (const float*)d_in[2];
    const float* W_ih    = (const float*)d_in[3];
    const float* W_hh    = (const float*)d_in[4];
    const float* b_ih    = (const float*)d_in[5];
    const float* b_hh    = (const float*)d_in[6];
    const float* W_score = (const float*)d_in[7];
    const float* b_score = (const float*)d_in[8];
    float* out = (float*)d_out;

    // --- workspace carve-up (256B aligned) ---
    char* ws = (char*)d_ws;
    size_t off = 0;
    auto carve = [&](size_t bytes) {
        size_t cur = off;
        off = (off + bytes + 255) & ~(size_t)255;
        return (void*)(ws + cur);
    };
    unsigned short* wfc_bf  = (unsigned short*)carve((size_t)Mh * Mh * 2);       // 8 MB
    unsigned short* wih_bf  = (unsigned short*)carve((size_t)G4 * Mh * 2);       // 32 MB
    unsigned short* whh_bf  = (unsigned short*)carve((size_t)G4 * Mh * 2);       // 32 MB
    unsigned short* pooled  = (unsigned short*)carve((size_t)Bb * Mh * 2);
    unsigned short* feat_bf = (unsigned short*)carve((size_t)Bb * Mh * 2);
    float*          xg      = (float*)carve((size_t)Bb * G4 * 4);                // 2 MB
    unsigned short* h_all   = (unsigned short*)carve((size_t)NINT * Bb * Mh * 2);// 26 MB
    float*          c_all   = (float*)carve((size_t)NINT * Bb * Mh * 4);         // 52 MB
    float*          o_buf   = (float*)carve((size_t)NINT * Bb * 4);
    (void)ws_size; (void)n_in; (void)in_sizes; (void)out_size;

    // --- host tree build (deterministic; recomputed every call) ---
    TreeDesc td;
    build_tree_host(td.fa, td.isLeft);
    int level[NINT];
    level[0] = 0;
    int maxLv = 0;
    for (int i = 1; i < NINT; ++i) {
        level[i] = level[td.fa[i]] + 1;
        if (level[i] > maxLv) maxLv = level[i];
    }

    // --- K2: weight converts ---
    {
        int n4 = Mh * Mh / 4;
        f2bf_kernel<<<(n4 + 255) / 256, 256, 0, stream>>>(W_fc, wfc_bf, n4);
        n4 = G4 * Mh / 4;
        f2bf_kernel<<<(n4 + 255) / 256, 256, 0, stream>>>(W_ih, wih_bf, n4);
        f2bf_kernel<<<(n4 + 255) / 256, 256, 0, stream>>>(W_hh, whh_bf, n4);
    }

    // --- K1: pool ---
    {
        int total = Bb * Mh;
        pool_kernel<<<(total + 255) / 256, 256, 0, stream>>>(x, pooled, total);
    }

    // --- K3a: feat = pooled @ W_fc^T + b_fc   -> bf16 ---
    gemm_wmma_bias<<<Mh / 128, 256, 0, stream>>>(
        pooled, wfc_bf, b_fc, nullptr, nullptr, feat_bf, Mh, Mh);

    // --- K3b: xg = feat @ W_ih^T + b_ih + b_hh -> fp32 ---
    gemm_wmma_bias<<<G4 / 128, 256, 0, stream>>>(
        feat_bf, wih_bf, b_ih, b_hh, xg, nullptr, G4, Mh);

    // --- K4: tree levels (fused W_hh GEMM + LSTM cell) ---
    for (int lv = 0; lv <= maxLv; ++lv) {
        LevelDesc ld;
        ld.count = 0;
        for (int i = 0; i < NINT; ++i)
            if (level[i] == lv) {
                ld.node[ld.count]   = i;
                ld.parent[ld.count] = td.fa[i];
                ld.count++;
            }
        lstm_level_kernel<<<dim3(Mh / 128, ld.count), 256, 0, stream>>>(
            whh_bf, xg, h_all, c_all, ld, lv == 0 ? 1 : 0);
    }

    // --- K5: per-node score dots ---
    {
        int pairs = NINT * Bb;                 // 6336 waves
        score_kernel<<<(pairs + 7) / 8, 256, 0, stream>>>(
            h_all, W_score, b_score, o_buf);
    }

    // --- K6: leaf logits ---
    {
        int total = Bb * NC;
        logit_kernel<<<(total + 255) / 256, 256, 0, stream>>>(o_buf, out, td);
    }
}
